// BiLevelGAT_2422361555840
// MI455X (gfx1250) — compile-verified, matrix operand order checked by reading, not run on hardware
//
#include <hip/hip_runtime.h>
#include <hip/hip_bf16.h>
#include <math.h>

// ---------------------------------------------------------------------------
// BiLevelGAT forward for MI455X (gfx1250, wave32, WMMA).
// Heavy GEMMs: double-buffered LDS tiling + V_WMMA_F32_16X16X32_F16.
// ---------------------------------------------------------------------------

typedef __attribute__((ext_vector_type(16))) _Float16 v16h;
typedef __attribute__((ext_vector_type(8)))  _Float16 v8h;
typedef __attribute__((ext_vector_type(8)))  float    v8f;

#define NU 4096
#define NI 4096
#define DD 256
#define DQ 4
#define BB 8192

#define BM 128
#define BN 64
#define BK 32
#define LDP (BK + 8)   // padded LDS row stride (40 halves = 80B -> no bank dup)

// ---------------------------------------------------------------------------
// Tiled GEMM: C[M,N] = act(A[M,K] @ B[K,N] + bias[N]), all f32 in global,
// f16 in LDS (converted once in the cooperative loader), f32 WMMA accumulate.
// Block: 256 threads = 8 waves arranged 4(M) x 2(N); wave tile 32x32 = 2x2
// WMMA 16x16 tiles. Requires M%128==0, N%64==0, K%32==0 (true for all uses).
// ---------------------------------------------------------------------------
template<bool HASBIAS, bool RELU>
__global__ __launch_bounds__(256)
void gemm_wmma_tiled(const float* __restrict__ A, const float* __restrict__ B,
                     const float* __restrict__ bias, float* __restrict__ C,
                     int M, int N, int K)
{
    __shared__ __align__(16) _Float16 As[2][BM][LDP];
    __shared__ __align__(16) _Float16 Bs[2][BN][LDP];   // transposed: [n][k]

    const int t     = threadIdx.x;
    const int tileM = blockIdx.y * BM;
    const int tileN = blockIdx.x * BN;

    // ---- cooperative staging mapping ----
    const int ar = t >> 1;              // A row within tile: 0..127
    const int ac = (t & 1) * 16;        // A col base: 0 or 16
    const int bn = t & 63;              // B col within tile: 0..63
    const int bk = (t >> 6) * 8;        // B k base: 0,8,16,24

    const float* Ag = A + (size_t)(tileM + ar) * K + ac;
    const float* Bg = B + (size_t)tileN + bn;

    float4 a4[4];
    float  breg[8];

    auto loadg = [&](int kk) {
        const float4* ap = (const float4*)(Ag + kk);
#pragma unroll
        for (int i = 0; i < 4; ++i) a4[i] = ap[i];
#pragma unroll
        for (int j = 0; j < 8; ++j) breg[j] = Bg[(size_t)(kk + bk + j) * N];
        // hint the next A tile toward the caches (global_prefetch_b8)
        __builtin_prefetch((const void*)(Ag + kk + BK), 0, 1);
    };

    auto store_lds = [&](int buf) {
        v8h h0, h1, hb;
#pragma unroll
        for (int i = 0; i < 4; ++i) {
            _Float16 x = (_Float16)a4[i].x, y = (_Float16)a4[i].y;
            _Float16 z = (_Float16)a4[i].z, w = (_Float16)a4[i].w;
            if (i < 2) { h0[4*i] = x; h0[4*i+1] = y; h0[4*i+2] = z; h0[4*i+3] = w; }
            else { h1[4*(i-2)] = x; h1[4*(i-2)+1] = y; h1[4*(i-2)+2] = z; h1[4*(i-2)+3] = w; }
        }
        *(v8h*)&As[buf][ar][ac]     = h0;
        *(v8h*)&As[buf][ar][ac + 8] = h1;
#pragma unroll
        for (int j = 0; j < 8; ++j) hb[j] = (_Float16)breg[j];
        *(v8h*)&Bs[buf][bn][bk] = hb;
    };

    // ---- per-wave compute mapping ----
    const int lane  = t & 31;
    const int wid   = t >> 5;
    const int waveM = wid >> 1;         // 0..3
    const int waveN = wid & 1;          // 0..1
    const int half  = lane >> 4;
    const int lr    = lane & 15;

    v8f acc[2][2] = {};

    auto compute = [&](int buf) {
        v16h afrag[2], bfrag[2];
#pragma unroll
        for (int mi = 0; mi < 2; ++mi) {
            const int row = waveM * 32 + mi * 16 + lr;
            // A 16x32 f16 layout: lane half 0 -> K 0..7 & 16..23, half 1 -> +8
            v8h c0 = *(const v8h*)&As[buf][row][half * 8];
            v8h c1 = *(const v8h*)&As[buf][row][16 + half * 8];
#pragma unroll
            for (int i = 0; i < 8; ++i) { afrag[mi][i] = c0[i]; afrag[mi][8 + i] = c1[i]; }
        }
#pragma unroll
        for (int ni = 0; ni < 2; ++ni) {
            const int col = waveN * 32 + ni * 16 + lr;
            v8h c0 = *(const v8h*)&Bs[buf][col][half * 8];
            v8h c1 = *(const v8h*)&Bs[buf][col][16 + half * 8];
#pragma unroll
            for (int i = 0; i < 8; ++i) { bfrag[ni][i] = c0[i]; bfrag[ni][8 + i] = c1[i]; }
        }
#pragma unroll
        for (int mi = 0; mi < 2; ++mi)
#pragma unroll
            for (int ni = 0; ni < 2; ++ni)
                acc[mi][ni] = __builtin_amdgcn_wmma_f32_16x16x32_f16(
                    false, afrag[mi], false, bfrag[ni],
                    (short)0, acc[mi][ni], false, false);
    };

    // ---- double-buffered main loop ----
    loadg(0);
    store_lds(0);
    __syncthreads();

    const int nsteps = K / BK;
    for (int s = 0; s < nsteps; ++s) {
        const int cur = s & 1;
        if (s + 1 < nsteps) loadg((s + 1) * BK);
        compute(cur);
        if (s + 1 < nsteps) store_lds(cur ^ 1);
        __syncthreads();
    }

    // ---- epilogue ----
#pragma unroll
    for (int ni = 0; ni < 2; ++ni) {
        const int col = tileN + waveN * 32 + ni * 16 + lr;
        const float bv = HASBIAS ? bias[col] : 0.f;
#pragma unroll
        for (int mi = 0; mi < 2; ++mi) {
#pragma unroll
            for (int r = 0; r < 8; ++r) {
                const int row = tileM + waveM * 32 + mi * 16 + half * 8 + r;
                float v = acc[mi][ni][r] + bv;
                if (RELU) v = fmaxf(v, 0.f);
                C[(size_t)row * N + col] = v;
            }
        }
    }
}

// ---------------------------------------------------------------------------
// out[i] = dot(X[i,:], v[:]) with D=256 -- one wave per row, float4 loads.
// ---------------------------------------------------------------------------
__global__ __launch_bounds__(256)
void rowdot_kernel(const float* __restrict__ X, const float* __restrict__ v,
                   float* __restrict__ out, int N)
{
    const int wave = (blockIdx.x * blockDim.x + threadIdx.x) >> 5;
    const int lane = threadIdx.x & 31;
    if (wave >= N) return;
    const float4* row4 = (const float4*)(X + (size_t)wave * DD);
    const float4* v4   = (const float4*)v;
    float4 p = row4[lane], a = v4[lane];
    float4 q = row4[lane + 32], b = v4[lane + 32];
    float s = p.x * a.x + p.y * a.y + p.z * a.z + p.w * a.w
            + q.x * b.x + q.y * b.y + q.z * b.z + q.w * b.w;
#pragma unroll
    for (int off = 16; off > 0; off >>= 1) s += __shfl_xor(s, off, 32);
    if (lane == 0) out[wave] = s;
}

// ---------------------------------------------------------------------------
// Masked-softmax attention row over Ndst=4096 columns.
// e = leaky_relu(a1[i] + a2[j], 0.2); empty rows -> all-zero alpha.
// 256-thread block per row; 16 contiguous cols per thread (int4/float4).
// ---------------------------------------------------------------------------
__global__ __launch_bounds__(256)
void gat_softmax(const float* __restrict__ a1, const float* __restrict__ a2,
                 const int* __restrict__ adj, float* __restrict__ alpha,
                 int Ndst)
{
    const int i = blockIdx.x;
    const int t = threadIdx.x;
    __shared__ float red[256];

    const float a1i  = a1[i];
    const int   base = t * 16;
    const int4*   adj4 = (const int4*)(adj + (size_t)i * Ndst + base);
    const float4* a24  = (const float4*)(a2 + base);

    float e[16];
    unsigned mbits = 0u;
    float lmax = -INFINITY;
#pragma unroll
    for (int qv = 0; qv < 4; ++qv) {
        const int4   m4 = adj4[qv];
        const float4 av = a24[qv];
        const float xs[4] = {av.x, av.y, av.z, av.w};
        const int   ms[4] = {m4.x, m4.y, m4.z, m4.w};
#pragma unroll
        for (int j2 = 0; j2 < 4; ++j2) {
            const int u = qv * 4 + j2;
            float x = a1i + xs[j2];
            x = (x > 0.f) ? x : 0.2f * x;      // leaky_relu 0.2
            e[u] = x;
            if (ms[j2] != 0) { mbits |= (1u << u); lmax = fmaxf(lmax, x); }
        }
    }

    red[t] = lmax;
    __syncthreads();
    for (int s = 128; s > 0; s >>= 1) {
        if (t < s) red[t] = fmaxf(red[t], red[t + s]);
        __syncthreads();
    }
    float m = red[0];
    if (m == -INFINITY) m = 0.f;               // guard empty rows
    __syncthreads();

    float ex[16];
    float lsum = 0.f;
#pragma unroll
    for (int u = 0; u < 16; ++u) {
        ex[u] = ((mbits >> u) & 1u) ? expf(e[u] - m) : 0.f;
        lsum += ex[u];
    }
    red[t] = lsum;
    __syncthreads();
    for (int s = 128; s > 0; s >>= 1) {
        if (t < s) red[t] += red[t + s];
        __syncthreads();
    }
    const float denom = red[0];
    const float inv = (denom > 0.f) ? 1.f / denom : 0.f;

    float4* out4 = (float4*)(alpha + (size_t)i * Ndst + base);
#pragma unroll
    for (int qv = 0; qv < 4; ++qv) {
        float4 o;
        o.x = ex[qv * 4 + 0] * inv;
        o.y = ex[qv * 4 + 1] * inv;
        o.z = ex[qv * 4 + 2] * inv;
        o.w = ex[qv * 4 + 3] * inv;
        out4[qv] = o;
    }
}

// ---------------------------------------------------------------------------
// h_u_out = elu(soc + cnt + h_u) + alpha_q * sigmoid(q_u@gate_W+gate_b)
//                                          * (q_u@qproj_W+qproj_b)
// ---------------------------------------------------------------------------
__global__ __launch_bounds__(256)
void combine_kernel(const float* __restrict__ soc, const float* __restrict__ cnt,
                    const float* __restrict__ hu, const float* __restrict__ qu,
                    const float* __restrict__ gateW, const float* __restrict__ gateb,
                    const float* __restrict__ qprojW, const float* __restrict__ qprojb,
                    const float* __restrict__ alpha_q, float* __restrict__ out)
{
    const int idx = blockIdx.x * blockDim.x + threadIdx.x;
    if (idx >= NU * DD) return;
    const int i = idx / DD;
    const int d = idx % DD;

    const float x = soc[idx] + cnt[idx] + hu[idx];
    const float h = (x > 0.f) ? x : expm1f(x);   // elu, alpha=1

    float g = gateb[d];
    float qp = qprojb[d];
#pragma unroll
    for (int k = 0; k < DQ; ++k) {
        const float q = qu[i * DQ + k];
        g  += q * gateW[k * DD + d];
        qp += q * qprojW[k * DD + d];
    }
    const float sg = 1.f / (1.f + expf(-g));
    out[idx] = h + alpha_q[0] * sg * qp;
}

// ---------------------------------------------------------------------------
// cat[b] = [h_u_out[user_idx[b]], h_i[item_idx[b]]]  (float4 granularity)
// ---------------------------------------------------------------------------
__global__ __launch_bounds__(256)
void gather_cat(const float* __restrict__ hu_out, const float* __restrict__ hi,
                const int* __restrict__ uidx, const int* __restrict__ iidx,
                float* __restrict__ cat)
{
    const int idx = blockIdx.x * blockDim.x + threadIdx.x;   // over B * 128
    if (idx >= BB * (2 * DD / 4)) return;
    const int b  = idx >> 7;          // / 128
    const int c4 = idx & 127;
    float4 v;
    if (c4 < 64) v = ((const float4*)(hu_out + (size_t)uidx[b] * DD))[c4];
    else         v = ((const float4*)(hi     + (size_t)iidx[b] * DD))[c4 - 64];
    ((float4*)cat)[idx] = v;
}

// ---------------------------------------------------------------------------
// r_hat[b] = 1 + 4*sigmoid(dot(hid[b], W2) + b2)  -- one wave per row.
// ---------------------------------------------------------------------------
__global__ __launch_bounds__(256)
void pred_final(const float* __restrict__ hid, const float* __restrict__ W2,
                const float* __restrict__ b2, float* __restrict__ out)
{
    const int wave = (blockIdx.x * blockDim.x + threadIdx.x) >> 5;
    const int lane = threadIdx.x & 31;
    if (wave >= BB) return;
    const float4* row4 = (const float4*)(hid + (size_t)wave * DD);
    const float4* w4   = (const float4*)W2;
    float4 p = row4[lane], a = w4[lane];
    float4 q = row4[lane + 32], b = w4[lane + 32];
    float s = p.x * a.x + p.y * a.y + p.z * a.z + p.w * a.w
            + q.x * b.x + q.y * b.y + q.z * b.z + q.w * b.w;
#pragma unroll
    for (int off = 16; off > 0; off >>= 1) s += __shfl_xor(s, off, 32);
    if (lane == 0) {
        const float r = s + b2[0];
        out[wave] = 1.f + 4.f / (1.f + expf(-r));
    }
}

// ---------------------------------------------------------------------------
// Host-side launch orchestration.
// ---------------------------------------------------------------------------
extern "C" void kernel_launch(void* const* d_in, const int* in_sizes, int n_in,
                              void* d_out, int out_size, void* d_ws, size_t ws_size,
                              hipStream_t stream)
{
    const float* x_u     = (const float*)d_in[0];
    const float* x_i     = (const float*)d_in[1];
    const int*   soc_adj = (const int*)  d_in[2];
    const int*   ui_adj  = (const int*)  d_in[3];
    const int*   uidx    = (const int*)  d_in[4];
    const int*   iidx    = (const int*)  d_in[5];
    const float* q_u     = (const float*)d_in[6];
    const float* user_W  = (const float*)d_in[7];
    const float* user_b  = (const float*)d_in[8];
    const float* item_W  = (const float*)d_in[9];
    const float* item_b  = (const float*)d_in[10];
    const float* soc_W   = (const float*)d_in[11];
    const float* soc_a   = (const float*)d_in[12];
    const float* cnt_W   = (const float*)d_in[13];
    const float* cnt_a   = (const float*)d_in[14];
    const float* gate_W  = (const float*)d_in[15];
    const float* gate_b  = (const float*)d_in[16];
    const float* qproj_W = (const float*)d_in[17];
    const float* qproj_b = (const float*)d_in[18];
    const float* alpha_q = (const float*)d_in[19];
    const float* pred_W1 = (const float*)d_in[20];
    const float* pred_b1 = (const float*)d_in[21];
    const float* pred_W2 = (const float*)d_in[22];
    const float* pred_b2 = (const float*)d_in[23];

    // outputs: r_hat [B], alpha_s [NU*NU], alpha_c [NU*NI]
    float* r_hat   = (float*)d_out;
    float* alpha_s = r_hat + BB;
    float* alpha_c = alpha_s + (size_t)NU * NU;

    // workspace layout (floats)
    float* ws = (float*)d_ws;
    size_t o = 0;
    float* h_u     = ws + o; o += (size_t)NU * DD;
    float* h_i     = ws + o; o += (size_t)NI * DD;
    float* Wh      = ws + o; o += (size_t)NU * DD;
    float* Wu      = ws + o; o += (size_t)NU * DD;
    float* Wi      = ws + o; o += (size_t)NI * DD;
    float* soc_msg = ws + o; o += (size_t)NU * DD;
    float* cnt_msg = ws + o; o += (size_t)NU * DD;
    float* hu_out  = ws + o; o += (size_t)NU * DD;
    float* a1s     = ws + o; o += NU;
    float* a2s     = ws + o; o += NU;
    float* a1c     = ws + o; o += NU;
    float* a2c     = ws + o; o += NI;
    float* cat     = ws + o; o += (size_t)BB * 2 * DD;
    float* hid     = ws + o; o += (size_t)BB * DD;

    const dim3 blk256(256);

    // --- input projections: h_u, h_i ---------------------------------------
    gemm_wmma_tiled<true, false><<<dim3(DD / BN, NU / BM), blk256, 0, stream>>>(
        x_u, user_W, user_b, h_u, NU, DD, 256);
    gemm_wmma_tiled<true, false><<<dim3(DD / BN, NI / BM), blk256, 0, stream>>>(
        x_i, item_W, item_b, h_i, NI, DD, 256);

    // --- attention input transforms: Wh, Wu, Wi ----------------------------
    gemm_wmma_tiled<false, false><<<dim3(DD / BN, NU / BM), blk256, 0, stream>>>(
        h_u, soc_W, nullptr, Wh, NU, DD, DD);
    gemm_wmma_tiled<false, false><<<dim3(DD / BN, NU / BM), blk256, 0, stream>>>(
        h_u, cnt_W, nullptr, Wu, NU, DD, DD);
    gemm_wmma_tiled<false, false><<<dim3(DD / BN, NI / BM), blk256, 0, stream>>>(
        h_i, cnt_W, nullptr, Wi, NI, DD, DD);

    // --- attention logits a1/a2 (rank-1 parts) -----------------------------
    rowdot_kernel<<<NU / 8, blk256, 0, stream>>>(Wh, soc_a,      a1s, NU);
    rowdot_kernel<<<NU / 8, blk256, 0, stream>>>(Wh, soc_a + DD, a2s, NU);
    rowdot_kernel<<<NU / 8, blk256, 0, stream>>>(Wu, cnt_a,      a1c, NU);
    rowdot_kernel<<<NI / 8, blk256, 0, stream>>>(Wi, cnt_a + DD, a2c, NI);

    // --- masked softmax -> alpha matrices (written straight into d_out) ----
    gat_softmax<<<NU, blk256, 0, stream>>>(a1s, a2s, soc_adj, alpha_s, NU);
    gat_softmax<<<NU, blk256, 0, stream>>>(a1c, a2c, ui_adj,  alpha_c, NI);

    // --- message aggregation: msg = alpha @ Wh_dst (K = 4096) --------------
    gemm_wmma_tiled<false, false><<<dim3(DD / BN, NU / BM), blk256, 0, stream>>>(
        alpha_s, Wh, nullptr, soc_msg, NU, DD, NU);
    gemm_wmma_tiled<false, false><<<dim3(DD / BN, NU / BM), blk256, 0, stream>>>(
        alpha_c, Wi, nullptr, cnt_msg, NU, DD, NI);

    // --- combine: elu + QoL gate -------------------------------------------
    combine_kernel<<<(NU * DD) / 256, blk256, 0, stream>>>(
        soc_msg, cnt_msg, h_u, q_u, gate_W, gate_b, qproj_W, qproj_b,
        alpha_q, hu_out);

    // --- prediction MLP -----------------------------------------------------
    gather_cat<<<(BB * 2 * DD / 4) / 256, blk256, 0, stream>>>(
        hu_out, h_i, uidx, iidx, cat);
    gemm_wmma_tiled<true, true><<<dim3(DD / BN, BB / BM), blk256, 0, stream>>>(
        cat, pred_W1, pred_b1, hid, BB, DD, 2 * DD);
    pred_final<<<(BB * 32) / 256, blk256, 0, stream>>>(
        hid, pred_W2, pred_b2, r_hat);
}
